// MemoryEfficientAttention_40003325395495
// MI455X (gfx1250) — matrix operand order repaired and checked by reading, not compiled, not run on hardware
//
#include <hip/hip_runtime.h>
#include <stdint.h>

typedef _Float16 f16;
typedef __attribute__((ext_vector_type(16))) _Float16 v16h;
typedef __attribute__((ext_vector_type(8)))  _Float16 v8h;
typedef __attribute__((ext_vector_type(8)))  float    v8f;

#define D_MODEL 1024
#define NHEADS  16
#define DK      64
#define BATCH   4
#define SEQ     2048
#define NTOK    (BATCH * SEQ)   // 8192

// -------- WMMA helpers (CDNA5 f16 16x16x32, f32 accumulate) --------

__device__ __forceinline__ v8f wmma_fp16(v16h a, v16h b, v8f c) {
  return __builtin_amdgcn_wmma_f32_16x16x32_f16(false, a, false, b, (short)0, c,
                                                false, false);
}

__device__ __forceinline__ v16h pack_frag(v8h lo, v8h hi) {
  v16h r;
#pragma unroll
  for (int i = 0; i < 8; ++i) { r[i] = lo[i]; r[8 + i] = hi[i]; }
  return r;
}

// A-fragment (16x32 f16): lane holds row M=lane%16.
//  elems i <-> k = kbase + half*16 + i
__device__ __forceinline__ v16h load_a_frag(const f16* row, int kbase, int half) {
  return pack_frag(*(const v8h*)(row + kbase + half * 16),
                   *(const v8h*)(row + kbase + half * 16 + 8));
}

// B-fragment (32x16 f16): lane holds col N=lane%16; elems i <-> k = half*16 + i,
// i.e. 16 contiguous f16 when stored column-of-B (k) major.
__device__ __forceinline__ v16h load_b_frag(const f16* p) {
  return pack_frag(*(const v8h*)p, *(const v8h*)(p + 8));
}

// -------- CDNA5 async global->LDS copy (16B per lane, ASYNCcnt) --------

__device__ __forceinline__ void async_copy_b128(const void* gsrc, void* ldst) {
  const unsigned lds = (unsigned)(uintptr_t)ldst;           // LDS = low 32 bits
  const unsigned long long g = (unsigned long long)(uintptr_t)gsrc;
  asm volatile("global_load_async_to_lds_b128 %0, %1, off"
               :
               : "v"(lds), "v"(g)
               : "memory");
}

__device__ __forceinline__ void wait_async0() {
  asm volatile("s_wait_asynccnt 0" ::: "memory");
}

// =====================================================================
// Kernel 0: weight pre-pass.  Wt[n][k] = (f16)W[k][n] for the 4 matrices.
// grid: (D/32, D/32, 4)  block: 256
// =====================================================================
__global__ void __launch_bounds__(256)
wtrans_kernel(const float* __restrict__ Wq, const float* __restrict__ Wk,
              const float* __restrict__ Wv, const float* __restrict__ Wo,
              f16* __restrict__ WtAll) {
  const int m = blockIdx.z;
  const float* W = (m == 0) ? Wq : (m == 1) ? Wk : (m == 2) ? Wv : Wo;
  f16* Wt = WtAll + (size_t)m * D_MODEL * D_MODEL;

  __shared__ f16 tile[32][33];
  const int tx = threadIdx.x & 31;
  const int ty0 = threadIdx.x >> 5;                 // 8 rows per pass
  const int k0 = blockIdx.x * 32, n0 = blockIdx.y * 32;

#pragma unroll
  for (int i = 0; i < 4; ++i) {
    const int ty = ty0 + i * 8;
    tile[ty][tx] = (f16)W[(size_t)(k0 + ty) * D_MODEL + n0 + tx];
  }
  __syncthreads();
#pragma unroll
  for (int i = 0; i < 4; ++i) {
    const int ty = ty0 + i * 8;
    Wt[(size_t)(n0 + ty) * D_MODEL + k0 + tx] = tile[tx][ty];
  }
}

// ------- async stage of a 64x32 f16 weight tile into padded LDS -------
// WtL: [64][40] f16 (pad 8 -> 80B row stride, 16B aligned, conflict-free)
__device__ __forceinline__ void gemm_stage_wt(const f16* __restrict__ WtG,
                                              int c0, int k0, int tid,
                                              f16 (*WtL)[40]) {
  // 64 rows x 64B tile = 256 chunks of 16B; 2 per thread, via async copies
#pragma unroll
  for (int i = 0; i < 2; ++i) {
    const int ch = tid + i * 128;
    const int n = ch >> 2, part = ch & 3;
    async_copy_b128(WtG + (size_t)(c0 + n) * D_MODEL + k0 + part * 8,
                    &WtL[n][part * 8]);
  }
}

// =====================================================================
// Kernel 1: fused Q/K/V projections.  Y = X @ W + b, fp32 in, f16 out.
// Q -> [B,H,S,DK] (pre-scaled 1/sqrt(dk));  K -> [B,H,S,DK];  V -> [B,H,DK,S]
// grid: (D/64, NTOK/64, 3)  block: 128 (4 waves x 16 rows)
// =====================================================================
__global__ void __launch_bounds__(128)
qkv_proj_kernel(const float* __restrict__ Xq, const float* __restrict__ Xk,
                const float* __restrict__ Xv, const f16* __restrict__ WtAll,
                const float* __restrict__ bq, const float* __restrict__ bk,
                const float* __restrict__ bv, f16* __restrict__ Qh,
                f16* __restrict__ Kh, f16* __restrict__ Vt) {
  const int proj = blockIdx.z;
  const float* X = (proj == 0) ? Xq : (proj == 1) ? Xk : Xv;
  const float* bias = (proj == 0) ? bq : (proj == 1) ? bk : bv;
  const f16* WtG = WtAll + (size_t)proj * D_MODEL * D_MODEL;

  const int c0 = blockIdx.x * 64;
  const int r0 = blockIdx.y * 64;
  const int tid = threadIdx.x;
  const int wave = tid >> 5;
  const int lane = tid & 31;
  const int half = lane >> 4;
  const int ln = lane & 15;

  __shared__ f16 WtL[64][40];

  const float* xrow = X + (size_t)(r0 + wave * 16 + ln) * D_MODEL;

  v8f acc[4] = {};
  for (int k0 = 0; k0 < D_MODEL; k0 += 32) {
    gemm_stage_wt(WtG, c0, k0, tid, WtL);
    wait_async0();
    __syncthreads();

    // A fragment: fp32 global -> f16 registers
    v16h afrag;
    {
      const float* p0 = xrow + k0 + half * 16;
      const float* p1 = xrow + k0 + half * 16 + 8;
#pragma unroll
      for (int i = 0; i < 8; ++i) {
        afrag[i] = (f16)p0[i];
        afrag[8 + i] = (f16)p1[i];
      }
    }
#pragma unroll
    for (int t = 0; t < 4; ++t) {
      v16h bfrag = load_b_frag(&WtL[t * 16 + ln][half * 16]);
      acc[t] = wmma_fp16(afrag, bfrag, acc[t]);
    }
    __syncthreads();
  }

  // ---- epilogue: block-constant head/batch, branch-free scatter ----
  // c = c0 + t*16 + ln with c0 % 64 == 0  ->  h = c0>>6, d = t*16+ln
  // r = r0 + (wave*16 + half*8 + j) with r0 % 64 == 0 -> b = r0>>11, s = s0+...
  const int hB = c0 >> 6;
  const int bB = r0 >> 11;
  const int s0 = (r0 & 2047) + wave * 16 + half * 8;
  const size_t headBase = ((size_t)(bB * NHEADS + hB)) * SEQ * DK;

  if (proj != 2) {
    f16* out = ((proj == 0) ? Qh : Kh) + headBase;
    const float scale = (proj == 0) ? 0.125f : 1.0f;  // 1/sqrt(64) folded in Q
#pragma unroll
    for (int t = 0; t < 4; ++t) {
      const int d = t * 16 + ln;
      const float bb = bias[c0 + d];
#pragma unroll
      for (int j = 0; j < 8; ++j) {
        out[(size_t)(s0 + j) * DK + d] = (f16)((acc[t][j] + bb) * scale);
      }
    }
  } else {
    f16* out = Vt + headBase;
#pragma unroll
    for (int t = 0; t < 4; ++t) {
      const int d = t * 16 + ln;
      const float bb = bias[c0 + d];
#pragma unroll
      for (int j = 0; j < 8; ++j) {
        out[(size_t)d * SEQ + s0 + j] = (f16)(acc[t][j] + bb);
      }
    }
  }
}

// =====================================================================
// Kernel 2: flash attention per (b,h).  O = softmax(Q K^T) V -> [B,S,D] f16
// grid: (SEQ/64, B*H)  block: 128 (4 waves x 16 q-rows)
// K/V chunks staged per block into LDS with async b128 copies.
// =====================================================================
__global__ void __launch_bounds__(128)
flash_attn_kernel(const f16* __restrict__ Qh, const f16* __restrict__ Kh,
                  const f16* __restrict__ Vt, f16* __restrict__ O) {
  const int bh = blockIdx.y;
  const int b = bh >> 4, h = bh & 15;
  const int tid = threadIdx.x;
  const int wave = tid >> 5;
  const int lane = tid & 31;
  const int half = lane >> 4;
  const int ln = lane & 15;

  const f16* Qbh = Qh + (size_t)bh * SEQ * DK;
  const f16* Kbh = Kh + (size_t)bh * SEQ * DK;
  const f16* Vbh = Vt + (size_t)bh * DK * SEQ;

  const int q0 = blockIdx.x * 64 + wave * 16;

  // Q fragments for this wave's 16 rows (dk=64 -> 2 chunks of 32)
  v16h qfrag[2];
  {
    const f16* qrow = Qbh + (size_t)(q0 + ln) * DK;
#pragma unroll
    for (int kc = 0; kc < 2; ++kc) qfrag[kc] = load_a_frag(qrow, kc * 32, half);
  }

  __shared__ f16 Kl[64][72];        // 64 keys x 64 dk (pad 8)
  __shared__ f16 Vl[64][72];        // 64 dk   x 64 keys (pad 8)
  __shared__ f16 Plds[4][16][72];   // per-wave P relayout tile

  v8f acc_o[4] = {};
  float m_i[8], l_i[8];
#pragma unroll
  for (int j = 0; j < 8; ++j) { m_i[j] = -1e30f; l_i[j] = 0.0f; }

  for (int kv0 = 0; kv0 < SEQ; kv0 += 64) {
    // ---- async-stage K and V chunks (8 KB each, 16B/lane copies) ----
#pragma unroll
    for (int i = 0; i < 4; ++i) {
      const int ch = tid + i * 128;               // 0..511
      const int n = ch >> 3, part = ch & 7;       // row, 16B slot
      async_copy_b128(Kbh + (size_t)(kv0 + n) * DK + part * 8, &Kl[n][part * 8]);
      async_copy_b128(Vbh + (size_t)n * SEQ + kv0 + part * 8, &Vl[n][part * 8]);
    }
    wait_async0();
    __syncthreads();

    // ---- scores S = (Q*scale) K^T ----
    v8f s_acc[4] = {};
#pragma unroll
    for (int t = 0; t < 4; ++t) {
#pragma unroll
      for (int kc = 0; kc < 2; ++kc) {
        v16h bfrag = load_b_frag(&Kl[t * 16 + ln][kc * 32 + half * 16]);
        s_acc[t] = wmma_fp16(qfrag[kc], bfrag, s_acc[t]);
      }
    }

    // ---- online softmax (rows m = half*8 + j) ----
    float pmax[8], corr[8], rsum[8];
#pragma unroll
    for (int j = 0; j < 8; ++j) {
      float v = fmaxf(fmaxf(s_acc[0][j], s_acc[1][j]),
                      fmaxf(s_acc[2][j], s_acc[3][j]));
#pragma unroll
      for (int m = 1; m < 16; m <<= 1) v = fmaxf(v, __shfl_xor(v, m, 32));
      pmax[j] = v;
    }
#pragma unroll
    for (int j = 0; j < 8; ++j) {
      const float mn = fmaxf(m_i[j], pmax[j]);
      corr[j] = __expf(m_i[j] - mn);
      m_i[j] = mn;
      rsum[j] = 0.0f;
    }
#pragma unroll
    for (int t = 0; t < 4; ++t) {
#pragma unroll
      for (int j = 0; j < 8; ++j) {
        const float p = __expf(s_acc[t][j] - m_i[j]);
        rsum[j] += p;
        Plds[wave][half * 8 + j][t * 16 + ln] = (f16)p;
      }
    }
#pragma unroll
    for (int j = 0; j < 8; ++j) {
      float v = rsum[j];
#pragma unroll
      for (int m = 1; m < 16; m <<= 1) v += __shfl_xor(v, m, 32);
      l_i[j] = l_i[j] * corr[j] + v;
#pragma unroll
      for (int t = 0; t < 4; ++t) acc_o[t][j] *= corr[j];
    }
    asm volatile("s_wait_dscnt 0" ::: "memory");  // P store->load (same wave)

    // ---- O += P @ V ----
    const f16* prow = &Plds[wave][ln][0];
#pragma unroll
    for (int kc = 0; kc < 2; ++kc) {
      v16h pfrag = load_a_frag(prow, kc * 32, half);
#pragma unroll
      for (int t = 0; t < 4; ++t) {
        v16h vfrag = load_b_frag(&Vl[t * 16 + ln][kc * 32 + half * 16]);
        acc_o[t] = wmma_fp16(pfrag, vfrag, acc_o[t]);
      }
    }
    __syncthreads();  // all waves done reading Kl/Vl before next stage
  }

  // ---- epilogue: O / l -> f16 [B,S,D] ----
  f16* obase = O + ((size_t)(b * SEQ + q0 + half * 8)) * D_MODEL + h * DK + ln;
#pragma unroll
  for (int t = 0; t < 4; ++t) {
#pragma unroll
    for (int j = 0; j < 8; ++j) {
      obase[(size_t)j * D_MODEL + t * 16] = (f16)(acc_o[t][j] / l_i[j]);
    }
  }
}

// =====================================================================
// Kernel 3: output projection.  Y = O @ Wo + bo, f16 in, fp32 out.
// grid: (D/64, NTOK/64)  block: 128
// =====================================================================
__global__ void __launch_bounds__(128)
out_proj_kernel(const f16* __restrict__ Oin, const f16* __restrict__ WtO,
                const float* __restrict__ bo, float* __restrict__ Y) {
  const int c0 = blockIdx.x * 64;
  const int r0 = blockIdx.y * 64;
  const int tid = threadIdx.x;
  const int wave = tid >> 5;
  const int lane = tid & 31;
  const int half = lane >> 4;
  const int ln = lane & 15;

  __shared__ f16 WtL[64][40];

  const f16* orow = Oin + (size_t)(r0 + wave * 16 + ln) * D_MODEL;

  v8f acc[4] = {};
  for (int k0 = 0; k0 < D_MODEL; k0 += 32) {
    gemm_stage_wt(WtO, c0, k0, tid, WtL);
    wait_async0();
    __syncthreads();

    v16h afrag = load_a_frag(orow, k0, half);
#pragma unroll
    for (int t = 0; t < 4; ++t) {
      v16h bfrag = load_b_frag(&WtL[t * 16 + ln][half * 16]);
      acc[t] = wmma_fp16(afrag, bfrag, acc[t]);
    }
    __syncthreads();
  }

  float* ybase = Y + (size_t)(r0 + wave * 16 + half * 8) * D_MODEL + c0 + ln;
#pragma unroll
  for (int t = 0; t < 4; ++t) {
    const float bb = bo[c0 + t * 16 + ln];
#pragma unroll
    for (int j = 0; j < 8; ++j) {
      ybase[(size_t)j * D_MODEL + t * 16] = acc[t][j] + bb;
    }
  }
}

// =====================================================================
extern "C" void kernel_launch(void* const* d_in, const int* in_sizes, int n_in,
                              void* d_out, int out_size, void* d_ws,
                              size_t ws_size, hipStream_t stream) {
  const float* q  = (const float*)d_in[0];
  const float* k  = (const float*)d_in[1];
  const float* v  = (const float*)d_in[2];
  const float* Wq = (const float*)d_in[3];
  const float* bq = (const float*)d_in[4];
  const float* Wk = (const float*)d_in[5];
  const float* bk = (const float*)d_in[6];
  const float* Wv = (const float*)d_in[7];
  const float* bv = (const float*)d_in[8];
  const float* Wo = (const float*)d_in[9];
  const float* bo = (const float*)d_in[10];

  // workspace: Wt (4x 1M f16 = 8 MB) | Qh | Kh | Vt | Oh (each 16 MB) = 72 MB
  const size_t welems = (size_t)D_MODEL * D_MODEL;
  const size_t elems = (size_t)NTOK * D_MODEL;
  f16* WtAll = (f16*)d_ws;
  f16* Qh = WtAll + 4 * welems;
  f16* Kh = Qh + elems;
  f16* Vt = Kh + elems;
  f16* Oh = Vt + elems;

  wtrans_kernel<<<dim3(D_MODEL / 32, D_MODEL / 32, 4), 256, 0, stream>>>(
      Wq, Wk, Wv, Wo, WtAll);
  qkv_proj_kernel<<<dim3(D_MODEL / 64, NTOK / 64, 3), 128, 0, stream>>>(
      q, k, v, WtAll, bq, bk, bv, Qh, Kh, Vt);
  flash_attn_kernel<<<dim3(SEQ / 64, BATCH * NHEADS), 128, 0, stream>>>(
      Qh, Kh, Vt, Oh);
  out_proj_kernel<<<dim3(D_MODEL / 64, NTOK / 64), 128, 0, stream>>>(
      Oh, WtAll + 3 * welems, bo, (float*)d_out);
}
